// SwinTransformerBlock_17652315587219
// MI455X (gfx1250) — matrix operand order repaired and checked by reading, not compiled, not run on hardware
//
#include <hip/hip_runtime.h>
#include <hip/hip_bf16.h>
#include <math.h>

// ---------------------------------------------------------------------------
// SwinV2 block for MI455X (gfx1250, wave32). All GEMMs via
// v_wmma_f32_16x16x32_bf16 (f32 accumulate). ~52 GFLOP/launch, ~300MB HBM
// traffic -> WMMA keeps it at the 23.3 TB/s bandwidth floor instead of
// f32-VALU-bound. 112KB dynamic LDS per WG (2 WGs per 320KB WGP).
// This rev: V stored transposed in LDS so every WMMA operand (A and B, all
// GEMMs) is fed by ds_load_b128 — no strided ds_load_u16 paths remain.
// ---------------------------------------------------------------------------

typedef __attribute__((ext_vector_type(16))) __bf16 v16bf;
typedef __attribute__((ext_vector_type(8)))  float  v8f;
typedef __attribute__((ext_vector_type(4)))  float  f32x4;
typedef __attribute__((ext_vector_type(4)))  unsigned int u32x4;

union BF16x16 { v16bf v; unsigned short u[16]; u32x4 q[2]; };

__device__ __forceinline__ unsigned short f2bf(float f) {
  __bf16 h = (__bf16)f;                       // v_cvt (RNE)
  return __builtin_bit_cast(unsigned short, h);
}
__device__ __forceinline__ float bf2f(unsigned short s) {
  __bf16 h = __builtin_bit_cast(__bf16, s);
  return (float)h;
}
__device__ __forceinline__ unsigned int pk2bf(float a, float b) {
  return (unsigned int)f2bf(a) | ((unsigned int)f2bf(b) << 16);
}

__device__ __forceinline__ v8f zero8() {
  v8f z = {0.f, 0.f, 0.f, 0.f, 0.f, 0.f, 0.f, 0.f};
  return z;
}
__device__ __forceinline__ v8f wmma_bf16(v16bf a, v16bf b, v8f c) {
  // (neg_a, A, neg_b, B, c_mod, C, reuse_a, reuse_b)
  return __builtin_amdgcn_wmma_f32_16x16x32_bf16(false, a, false, b, (short)0, c,
                                                 false, false);
}

// A fragment, 16x32 (MxK), source row-major src[row][k], leading dim ld.
// ISA layout: lane L<16 -> row L, K in {base..base+7} U {base+16..base+23},
// base = (lane>>4)*8. Contiguous 8-halves -> two ds_load_b128.
// Requires (k0 + base) % 8 == 0 and ld % 8 == 0 (true at all call sites).
__device__ __forceinline__ v16bf load_a_bf(const unsigned short* src, int row0,
                                           int k0, int ld, int lane) {
  BF16x16 r;
  const u32x4* p = (const u32x4*)(src + (row0 + (lane & 15)) * ld + k0 +
                                  ((lane >> 4) << 3));
  r.q[0] = p[0];       // K base..base+7
  r.q[1] = p[2];       // K base+16..base+23
  return r.v;
}
__device__ __forceinline__ v16bf load_a_f32(const float* src, int row0, int k0,
                                            int ld, int lane) {
  BF16x16 r;
  const f32x4* p = (const f32x4*)(src + (row0 + (lane & 15)) * ld + k0 +
                                  ((lane >> 4) << 3));
  f32x4 a0 = p[0], a1 = p[1], b0 = p[4], b1 = p[5];
#pragma unroll
  for (int e = 0; e < 4; ++e) { r.u[e] = f2bf(a0[e]); r.u[4 + e] = f2bf(a1[e]); }
#pragma unroll
  for (int e = 0; e < 4; ++e) { r.u[8 + e] = f2bf(b0[e]); r.u[12 + e] = f2bf(b1[e]); }
  return r.v;
}
// B fragment 32x16 (KxN) where B[k][n] = src[n][k] (src row-major over n).
// ISA layout: col n = lane&15, K in {base..base+7} U {base+16..base+23},
// base = (lane>>4)*8 -> two ds_load_b128. Requires k0 % 16 == 0 and ld % 16 == 0 (true here).
__device__ __forceinline__ v16bf load_bT_bf(const unsigned short* src, int k0,
                                            int col0, int ld, int lane) {
  BF16x16 r;
  const u32x4* p = (const u32x4*)(src + (col0 + (lane & 15)) * ld + k0 +
                                  ((lane >> 4) << 3));
  r.q[0] = p[0];
  r.q[1] = p[2];
  return r.v;
}
__device__ __forceinline__ v16bf load_bT_gf32(const float* src, int k0, int col0,
                                              int ld, int lane) {
  BF16x16 r;
  const f32x4* p = (const f32x4*)(src + (col0 + (lane & 15)) * ld + k0 +
                                  ((lane >> 4) << 3));
#pragma unroll
  for (int c = 0; c < 4; ++c) {
    f32x4 v = p[(c & 1) + ((c >> 1) << 2)];
#pragma unroll
    for (int e = 0; e < 4; ++e) r.u[c * 4 + e] = f2bf(v[e]);
  }
  return r.v;
}

// region code for the 3D shift mask (WS=4, SS=2, dim 32): regions split at 28,30
__device__ __forceinline__ int regcode3(int wd, int wh, int ww, int n) {
  int pd = (wd << 2) + (n >> 4);
  int ph = (wh << 2) + ((n >> 2) & 3);
  int pw = (ww << 2) + (n & 3);
  int rd = (pd < 28) ? 0 : ((pd < 30) ? 1 : 2);
  int rh = (ph < 28) ? 0 : ((ph < 30) ? 1 : 2);
  int rw = (pw < 28) ? 0 : ((pw < 30) ? 1 : 2);
  return rd * 9 + rh * 3 + rw;
}

// ---------------------------------------------------------------------------
// Kernel 0: CPB MLP -> rpb[4][64][64] = 16*sigmoid(...), scale[4]
// ---------------------------------------------------------------------------
__global__ void __launch_bounds__(256) swin_cpb_kernel(
    const float* __restrict__ cpb_w1, const float* __restrict__ cpb_b1,
    const float* __restrict__ cpb_w2, const float* __restrict__ logit_scale,
    float* __restrict__ rpb, float* __restrict__ scale) {
  __shared__ float btab[343 * 4];
  int tid = threadIdx.x;
  for (int t = tid; t < 343; t += 256) {
    int i = t / 49, j = (t / 7) % 7, k = t % 7;
    float c[3] = {(float)(i - 3), (float)(j - 3), (float)(k - 3)};
#pragma unroll
    for (int d = 0; d < 3; ++d) {
      float v = c[d] * (8.0f / 3.0f);
      float sg = (v > 0.f) ? 1.f : ((v < 0.f) ? -1.f : 0.f);
      c[d] = sg * log2f(fabsf(v) + 1.0f) * (1.0f / 3.0f);  // /log2(8)
    }
    float acc[4] = {0.f, 0.f, 0.f, 0.f};
    for (int h = 0; h < 512; ++h) {
      float hv = c[0] * cpb_w1[h * 3 + 0] + c[1] * cpb_w1[h * 3 + 1] +
                 c[2] * cpb_w1[h * 3 + 2] + cpb_b1[h];
      hv = fmaxf(hv, 0.f);
#pragma unroll
      for (int q = 0; q < 4; ++q) acc[q] += hv * cpb_w2[q * 512 + h];
    }
#pragma unroll
    for (int q = 0; q < 4; ++q) btab[t * 4 + q] = acc[q];
  }
  __syncthreads();
  for (int e = tid; e < 4 * 64 * 64; e += 256) {
    int h = e >> 12, n = (e >> 6) & 63, m = e & 63;
    int d0 = (n >> 4) - (m >> 4) + 3;
    int d1 = ((n >> 2) & 3) - ((m >> 2) & 3) + 3;
    int d2 = (n & 3) - (m & 3) + 3;
    float b = btab[(d0 * 49 + d1 * 7 + d2) * 4 + h];
    rpb[e] = 16.0f / (1.0f + expf(-b));
  }
  if (tid < 4) scale[tid] = expf(fminf(logit_scale[tid], 4.60517018599f));
}

// ---------------------------------------------------------------------------
// Kernel 1: fused windowed attention (one window / workgroup, 8 waves)
// LDS arena (112KB): Q 0..16K | K 16..32K | VT 32..48K (transposed, ld=64) |
//                    XW/S 48..64K | O 64..80K | PW 80..112K
//                    PO(f32) reuses Q+K at 0..32K
// ---------------------------------------------------------------------------
__global__ void __launch_bounds__(256) swin_attn_kernel(
    const float* __restrict__ x, const float* __restrict__ qkv_w,
    const float* __restrict__ q_bias, const float* __restrict__ v_bias,
    const float* __restrict__ proj_w, const float* __restrict__ proj_b,
    const float* __restrict__ n1g, const float* __restrict__ n1b,
    const float* __restrict__ rpb, const float* __restrict__ scale,
    float* __restrict__ x1) {
  extern __shared__ char smem[];
  unsigned short* Q  = (unsigned short*)(smem);
  unsigned short* K  = (unsigned short*)(smem + 16384);
  unsigned short* VT = (unsigned short*)(smem + 32768);  // [c][token], ld=64
  unsigned short* XW = (unsigned short*)(smem + 49152);
  float*          S  = (float*)(smem + 49152);
  unsigned short* O  = (unsigned short*)(smem + 65536);
  unsigned short* PW = (unsigned short*)(smem + 81920);
  float*          PO = (float*)(smem);

  const int tid = threadIdx.x, lane = tid & 31, wave = tid >> 5;
  const int wi = blockIdx.x;
  const int b = wi >> 9, w = wi & 511;
  const int wd = w >> 6, wh = (w >> 3) & 7, ww = w & 7;

  // gather shifted window (roll -2 on D,H only) -> bf16 LDS, float4 reads
  for (int i = tid; i < 64 * 32; i += 256) {
    int n = i >> 5, c4 = (i & 31) << 2;
    int d  = ((wd << 2) + (n >> 4) + 2) & 31;
    int hh = ((wh << 2) + ((n >> 2) & 3) + 2) & 31;
    int wp = (ww << 2) + (n & 3);
    int src = (b << 15) + (d << 10) + (hh << 5) + wp;
    f32x4 v = *(const f32x4*)(x + src * 128 + c4);
    unsigned int* dst = (unsigned int*)(XW + n * 128 + c4);
    dst[0] = pk2bf(v.x, v.y);
    dst[1] = pk2bf(v.z, v.w);
  }
  __syncthreads();

  // ---- QKV: (64x128)@(128x384), B from global qkv_w (o-major = transposed)
  for (int ct = wave * 3; ct < wave * 3 + 3; ++ct) {
    v8f acc[4];
#pragma unroll
    for (int t = 0; t < 4; ++t) acc[t] = zero8();
#pragma unroll
    for (int ks = 0; ks < 4; ++ks) {
      v16bf bf = load_bT_gf32(qkv_w, ks * 32, ct * 16, 128, lane);
#pragma unroll
      for (int rt = 0; rt < 4; ++rt) {
        v16bf af = load_a_bf(XW, rt * 16, ks * 32, 128, lane);
        acc[rt] = wmma_bf16(af, bf, acc[rt]);
      }
    }
    int o = ct * 16 + (lane & 15);
    float bias = (o < 128) ? q_bias[o] : ((o >= 256) ? v_bias[o - 256] : 0.f);
    int cc = o & 127;
    if (o < 256) {  // q, k: token-major [row][c]
      unsigned short* dst = (o < 128) ? Q : K;
#pragma unroll
      for (int rt = 0; rt < 4; ++rt)
#pragma unroll
        for (int r = 0; r < 8; ++r) {
          int row = rt * 16 + ((lane >> 4) << 3) + r;
          dst[row * 128 + cc] = f2bf(acc[rt][r] + bias);
        }
    } else {        // v: transposed [c][token] so attn@v B loads are b128
#pragma unroll
      for (int rt = 0; rt < 4; ++rt)
#pragma unroll
        for (int r = 0; r < 8; ++r) {
          int row = rt * 16 + ((lane >> 4) << 3) + r;
          VT[cc * 64 + row] = f2bf(acc[rt][r] + bias);
        }
    }
  }
  __syncthreads();

  // ---- cosine normalize q (fold per-head logit scale in) and k
  {
    int n = tid >> 2, h = tid & 3;
    unsigned short* qr = Q + n * 128 + h * 32;
    unsigned short* kr = K + n * 128 + h * 32;
    float sq = 0.f, sk = 0.f;
#pragma unroll
    for (int c = 0; c < 32; ++c) { float a = bf2f(qr[c]); sq += a * a; }
#pragma unroll
    for (int c = 0; c < 32; ++c) { float a = bf2f(kr[c]); sk += a * a; }
    float iq = scale[h] / fmaxf(sqrtf(sq), 1e-12f);
    float ik = 1.0f    / fmaxf(sqrtf(sk), 1e-12f);
#pragma unroll
    for (int c = 0; c < 32; ++c) qr[c] = f2bf(bf2f(qr[c]) * iq);
#pragma unroll
    for (int c = 0; c < 32; ++c) kr[c] = f2bf(bf2f(kr[c]) * ik);
  }
  __syncthreads();

  // ---- per-head: attn = qn@kn^T + rpb + mask ; softmax ; attn@v
  for (int h = 0; h < 4; ++h) {
#pragma unroll
    for (int tt = 0; tt < 2; ++tt) {
      int t = wave * 2 + tt;
      int rt = t >> 2, ct = t & 3;
      v16bf af = load_a_bf(Q, rt * 16, h * 32, 128, lane);
      v16bf bf = load_bT_bf(K, h * 32, ct * 16, 128, lane);
      v8f acc = wmma_bf16(af, bf, zero8());
      int col = ct * 16 + (lane & 15);
      int regc = regcode3(wd, wh, ww, col);
      const float* rp = rpb + (h << 12);
#pragma unroll
      for (int r = 0; r < 8; ++r) {
        int row = rt * 16 + ((lane >> 4) << 3) + r;
        float mval = (regcode3(wd, wh, ww, row) != regc) ? -100.0f : 0.0f;
        S[row * 64 + col] = acc[r] + rp[row * 64 + col] + mval;
      }
    }
    __syncthreads();
    {  // softmax: 4 lanes per row, reduce within aligned lane quads
      int row = tid >> 2, part = tid & 3;
      float* sr = S + row * 64 + part * 16;
      float mx = -3.0e38f;
#pragma unroll
      for (int j = 0; j < 16; ++j) mx = fmaxf(mx, sr[j]);
      mx = fmaxf(mx, __shfl_xor(mx, 1, 32));
      mx = fmaxf(mx, __shfl_xor(mx, 2, 32));
      float ev[16]; float sum = 0.f;
#pragma unroll
      for (int j = 0; j < 16; ++j) { ev[j] = expf(sr[j] - mx); sum += ev[j]; }
      sum += __shfl_xor(sum, 1, 32);
      sum += __shfl_xor(sum, 2, 32);
      float inv = 1.0f / sum;
#pragma unroll
      for (int j = 0; j < 16; ++j) sr[j] = ev[j] * inv;
    }
    __syncthreads();
    {  // out_h = attn(64x64) @ v_h(64x32), one tile per wave
      int rt = wave >> 1, ct = wave & 1;
      v8f acc = zero8();
#pragma unroll
      for (int ks = 0; ks < 2; ++ks) {
        v16bf af = load_a_f32(S, rt * 16, ks * 32, 64, lane);
        // B[k=token][n=c] = VT[h*32+ct*16+n][token k] -> transposed pattern
        v16bf bf = load_bT_bf(VT, ks * 32, h * 32 + ct * 16, 64, lane);
        acc = wmma_bf16(af, bf, acc);
      }
      int col = h * 32 + ct * 16 + (lane & 15);
#pragma unroll
      for (int r = 0; r < 8; ++r) {
        int row = rt * 16 + ((lane >> 4) << 3) + r;
        O[row * 128 + col] = f2bf(acc[r]);
      }
    }
    __syncthreads();
  }

  // ---- proj: (64x128)@(128x128); stage proj_w verbatim (c-major) as bf16
  for (int i = tid; i < 128 * 32; i += 256) {
    int r = i >> 5, c4 = (i & 31) << 2;
    f32x4 v = *(const f32x4*)(proj_w + r * 128 + c4);
    unsigned int* dst = (unsigned int*)(PW + r * 128 + c4);
    dst[0] = pk2bf(v.x, v.y);
    dst[1] = pk2bf(v.z, v.w);
  }
  __syncthreads();
  {
    int ct = wave;
    v8f acc[4];
#pragma unroll
    for (int t = 0; t < 4; ++t) acc[t] = zero8();
#pragma unroll
    for (int ks = 0; ks < 4; ++ks) {
      v16bf bf = load_bT_bf(PW, ks * 32, ct * 16, 128, lane);
#pragma unroll
      for (int rt = 0; rt < 4; ++rt) {
        v16bf af = load_a_bf(O, rt * 16, ks * 32, 128, lane);
        acc[rt] = wmma_bf16(af, bf, acc[rt]);
      }
    }
    int col = ct * 16 + (lane & 15);
    float pb = proj_b[col];
#pragma unroll
    for (int rt = 0; rt < 4; ++rt)
#pragma unroll
      for (int r = 0; r < 8; ++r) {
        int row = rt * 16 + ((lane >> 4) << 3) + r;
        PO[row * 128 + col] = acc[rt][r] + pb;
      }
  }
  __syncthreads();

  // ---- x1 = shortcut + LN(proj_out), scattered via inverse roll
  for (int row = wave; row < 64; row += 8) {
    const float* pr = PO + row * 128;
    int c0 = lane * 4;
    f32x4 v = *(const f32x4*)(pr + c0);
    float s = v.x + v.y + v.z + v.w;
#pragma unroll
    for (int off = 1; off < 32; off <<= 1) s += __shfl_xor(s, off, 32);
    float mean = s * (1.0f / 128.0f);
    float d0 = v.x - mean, d1 = v.y - mean, d2 = v.z - mean, d3 = v.w - mean;
    float vv = d0 * d0 + d1 * d1 + d2 * d2 + d3 * d3;
#pragma unroll
    for (int off = 1; off < 32; off <<= 1) vv += __shfl_xor(vv, off, 32);
    float inv = rsqrtf(vv * (1.0f / 128.0f) + 1e-5f);
    int d  = ((wd << 2) + (row >> 4) + 2) & 31;
    int hh = ((wh << 2) + ((row >> 2) & 3) + 2) & 31;
    int wp = (ww << 2) + (row & 3);
    int g = ((b << 15) + (d << 10) + (hh << 5) + wp) * 128 + c0;
    x1[g + 0] = x[g + 0] + d0 * inv * n1g[c0 + 0] + n1b[c0 + 0];
    x1[g + 1] = x[g + 1] + d1 * inv * n1g[c0 + 1] + n1b[c0 + 1];
    x1[g + 2] = x[g + 2] + d2 * inv * n1g[c0 + 2] + n1b[c0 + 2];
    x1[g + 3] = x[g + 3] + d3 * inv * n1g[c0 + 3] + n1b[c0 + 3];
  }
}

// ---------------------------------------------------------------------------
// Kernel 2: fused MLP (fc1+GELU+fc2) + post-LN residual, in-place on x1
// LDS: XB 0..16K | HB 16..80K (f32 PO reuses 16..48K) | WB 80..112K
// ---------------------------------------------------------------------------
__global__ void __launch_bounds__(256) swin_mlp_kernel(
    const float* __restrict__ fc1_w, const float* __restrict__ fc1_b,
    const float* __restrict__ fc2_w, const float* __restrict__ fc2_b,
    const float* __restrict__ n2g, const float* __restrict__ n2b,
    float* __restrict__ xio) {
  extern __shared__ char smem[];
  unsigned short* XB = (unsigned short*)(smem);
  unsigned short* HB = (unsigned short*)(smem + 16384);
  float*          PO = (float*)(smem + 16384);
  unsigned short* WB = (unsigned short*)(smem + 81920);
  const int tid = threadIdx.x, lane = tid & 31, wave = tid >> 5;
  const int row0 = blockIdx.x * 64;
  const int ct = wave;

  for (int i = tid; i < 64 * 32; i += 256) {
    f32x4 v = *(const f32x4*)(xio + row0 * 128 + i * 4);
    unsigned int* dst = (unsigned int*)(XB + i * 4);
    dst[0] = pk2bf(v.x, v.y);
    dst[1] = pk2bf(v.z, v.w);
  }
  __syncthreads();

  // fc1 + exact GELU -> HB (64x512 bf16), 128-wide output panels
  for (int ob = 0; ob < 4; ++ob) {
    for (int i = tid; i < 128 * 32; i += 256) {
      f32x4 v = *(const f32x4*)(fc1_w + ob * 16384 + i * 4);  // native (o,k)
      unsigned int* dst = (unsigned int*)(WB + i * 4);
      dst[0] = pk2bf(v.x, v.y);
      dst[1] = pk2bf(v.z, v.w);
      if (ob < 3) __builtin_prefetch(fc1_w + (ob + 1) * 16384 + i * 4, 0, 1);
    }
    __syncthreads();
    v8f acc[4];
#pragma unroll
    for (int t = 0; t < 4; ++t) acc[t] = zero8();
#pragma unroll
    for (int ks = 0; ks < 4; ++ks) {
      v16bf bf = load_bT_bf(WB, ks * 32, ct * 16, 128, lane);
#pragma unroll
      for (int rt = 0; rt < 4; ++rt) {
        v16bf af = load_a_bf(XB, rt * 16, ks * 32, 128, lane);
        acc[rt] = wmma_bf16(af, bf, acc[rt]);
      }
    }
    int o = ob * 128 + ct * 16 + (lane & 15);
    float bias = fc1_b[o];
#pragma unroll
    for (int rt = 0; rt < 4; ++rt)
#pragma unroll
      for (int r = 0; r < 8; ++r) {
        int row = rt * 16 + ((lane >> 4) << 3) + r;
        float hv = acc[rt][r] + bias;
        hv = 0.5f * hv * (1.0f + erff(hv * 0.70710678118f));  // exact GELU
        HB[row * 512 + o] = f2bf(hv);
      }
    __syncthreads();
  }

  // fc2: out(64x128), K=512 accumulated over 4 staged weight panels
  v8f acc[4];
#pragma unroll
  for (int t = 0; t < 4; ++t) acc[t] = zero8();
  for (int kb = 0; kb < 4; ++kb) {
    for (int i = tid; i < 128 * 32; i += 256) {
      int c = i >> 5, o4 = (i & 31) << 2;
      f32x4 v = *(const f32x4*)(fc2_w + c * 512 + kb * 128 + o4);
      unsigned int* dst = (unsigned int*)(WB + c * 128 + o4);  // (c, o_local)
      dst[0] = pk2bf(v.x, v.y);
      dst[1] = pk2bf(v.z, v.w);
    }
    __syncthreads();
#pragma unroll
    for (int ks = 0; ks < 4; ++ks) {
      v16bf bf = load_bT_bf(WB, ks * 32, ct * 16, 128, lane);
#pragma unroll
      for (int rt = 0; rt < 4; ++rt) {
        v16bf af = load_a_bf(HB, rt * 16, kb * 128 + ks * 32, 512, lane);
        acc[rt] = wmma_bf16(af, bf, acc[rt]);
      }
    }
    __syncthreads();
  }
  {
    int col = ct * 16 + (lane & 15);
    float bias = fc2_b[col];
#pragma unroll
    for (int rt = 0; rt < 4; ++rt)
#pragma unroll
      for (int r = 0; r < 8; ++r) {
        int row = rt * 16 + ((lane >> 4) << 3) + r;
        PO[row * 128 + col] = acc[rt][r] + bias;
      }
  }
  __syncthreads();

  // x = x1 + LN(h)
  for (int row = wave; row < 64; row += 8) {
    const float* pr = PO + row * 128;
    int c0 = lane * 4;
    f32x4 v = *(const f32x4*)(pr + c0);
    float s = v.x + v.y + v.z + v.w;
#pragma unroll
    for (int off = 1; off < 32; off <<= 1) s += __shfl_xor(s, off, 32);
    float mean = s * (1.0f / 128.0f);
    float d0 = v.x - mean, d1 = v.y - mean, d2 = v.z - mean, d3 = v.w - mean;
    float vv = d0 * d0 + d1 * d1 + d2 * d2 + d3 * d3;
#pragma unroll
    for (int off = 1; off < 32; off <<= 1) vv += __shfl_xor(vv, off, 32);
    float inv = rsqrtf(vv * (1.0f / 128.0f) + 1e-5f);
    int g = (row0 + row) * 128 + c0;
    xio[g + 0] = xio[g + 0] + d0 * inv * n2g[c0 + 0] + n2b[c0 + 0];
    xio[g + 1] = xio[g + 1] + d1 * inv * n2g[c0 + 1] + n2b[c0 + 1];
    xio[g + 2] = xio[g + 2] + d2 * inv * n2g[c0 + 2] + n2b[c0 + 2];
    xio[g + 3] = xio[g + 3] + d3 * inv * n2g[c0 + 3] + n2b[c0 + 3];
  }
}

// ---------------------------------------------------------------------------
extern "C" void kernel_launch(void* const* d_in, const int* in_sizes, int n_in,
                              void* d_out, int out_size, void* d_ws, size_t ws_size,
                              hipStream_t stream) {
  const float* x    = (const float*)d_in[0];
  const float* qkvw = (const float*)d_in[1];
  const float* qb   = (const float*)d_in[2];
  const float* vb   = (const float*)d_in[3];
  const float* ls   = (const float*)d_in[4];
  const float* w1   = (const float*)d_in[5];
  const float* b1   = (const float*)d_in[6];
  const float* w2   = (const float*)d_in[7];
  const float* pw   = (const float*)d_in[8];
  const float* pb   = (const float*)d_in[9];
  const float* g1   = (const float*)d_in[10];
  const float* bb1  = (const float*)d_in[11];
  const float* g2   = (const float*)d_in[12];
  const float* bb2  = (const float*)d_in[13];
  const float* f1w  = (const float*)d_in[14];
  const float* f1b  = (const float*)d_in[15];
  const float* f2w  = (const float*)d_in[16];
  const float* f2b  = (const float*)d_in[17];
  float* out = (float*)d_out;

  float* rpb   = (float*)d_ws;           // 4*64*64 f32 = 64KB
  float* scale = rpb + 4 * 64 * 64;      // 4 f32

  swin_cpb_kernel<<<1, 256, 0, stream>>>(w1, b1, w2, ls, rpb, scale);
  // 2048 windows (B=4 * 512 windows/batch), one per WG; d_out holds x1
  swin_attn_kernel<<<2048, 256, 112 * 1024, stream>>>(
      x, qkvw, qb, vb, pw, pb, g1, bb1, rpb, scale, out);
  // 131072 rows / 64 per WG; in-place on d_out
  swin_mlp_kernel<<<2048, 256, 112 * 1024, stream>>>(
      f1w, f1b, f2w, f2b, g2, bb2, out);
}
